// GraphAttention_32547262169799
// MI455X (gfx1250) — matrix operand order repaired and checked by reading, not compiled, not run on hardware
//
#include <hip/hip_runtime.h>
#include <hip/hip_bf16.h>

// CDNA5 / gfx1250 graph-attention. All matmuls via v_wmma_f32_16x16x32_bf16.
typedef __attribute__((ext_vector_type(16))) __bf16 v16bf;
typedef __attribute__((ext_vector_type(8)))  float  v8f;

#define HEADS 8
#define DH    64
#define NN    1024
#define DIM   256
#define INNER 512
#define SCALE_F 0.125f   // 64^-0.5

// Build an A/B fragment (v16bf) from two contiguous 8-element bf16 runs
// (elements 0..7 from p0, 8..15 from p1) using two 16B loads, no unpack VALU.
static __device__ __forceinline__ v16bf frag_bf16(const __bf16* p0,
                                                  const __bf16* p1) {
  union { uint4 u[2]; v16bf v; } t;
  t.u[0] = *(const uint4*)p0;
  t.u[1] = *(const uint4*)p1;
  return t.v;
}

// Same but from f32 memory (vectorized float4 loads + cvt to bf16).
static __device__ __forceinline__ v16bf frag_f32(const float* p0,
                                                 const float* p1) {
  float4 a = *(const float4*)p0, b = *(const float4*)(p0 + 4);
  float4 c = *(const float4*)p1, d = *(const float4*)(p1 + 4);
  v16bf v;
  v[0]  = (__bf16)a.x; v[1]  = (__bf16)a.y; v[2]  = (__bf16)a.z; v[3]  = (__bf16)a.w;
  v[4]  = (__bf16)b.x; v[5]  = (__bf16)b.y; v[6]  = (__bf16)b.z; v[7]  = (__bf16)b.w;
  v[8]  = (__bf16)c.x; v[9]  = (__bf16)c.y; v[10] = (__bf16)c.z; v[11] = (__bf16)c.w;
  v[12] = (__bf16)d.x; v[13] = (__bf16)d.y; v[14] = (__bf16)d.z; v[15] = (__bf16)d.w;
  return v;
}

// ---------------------------------------------------------------------------
// Kernel 1: fused P = nodes @ [Wq | Wk | Wv | Wg]  (M=2048, K=256, N=2048)
// Each wave: 32x16 output (2 M-tiles sharing B fragments).
// ---------------------------------------------------------------------------
__global__ __launch_bounds__(256) void qkvg_proj(
    const float* __restrict__ nodes, const float* __restrict__ Wq,
    const float* __restrict__ Wkv,   const float* __restrict__ Wg,
    __bf16* __restrict__ Qb, __bf16* __restrict__ Kb,
    __bf16* __restrict__ Vb, float* __restrict__ gate)
{
  int wave  = threadIdx.x >> 5;
  int lane  = threadIdx.x & 31;
  int tile  = blockIdx.x * 8 + wave;        // 8192 tiles = 64 mt2 x 128 nt
  int mt2   = tile >> 7;
  int nt    = tile & 127;
  int khalf = lane >> 4;
  int lrow  = lane & 15;
  int m0    = mt2 * 32 + lrow;
  int m1    = m0 + 16;

  int sec   = (nt * 16) >> 9;               // 0=q 1=k 2=v 3=g (wave-uniform)
  int cbase = (nt * 16) & 511;
  const float* W; int Nw; int coff;
  if      (sec == 0) { W = Wq;  Nw = 512;  coff = cbase;       }
  else if (sec == 1) { W = Wkv; Nw = 1024; coff = cbase;       }
  else if (sec == 2) { W = Wkv; Nw = 1024; coff = cbase + 512; }
  else               { W = Wg;  Nw = 512;  coff = cbase;       }
  int bc = coff + lrow;

  v8f acc0 = {}, acc1 = {};
  for (int kk = 0; kk < 8; ++kk) {          // K = 256 = 8 x 32
    const float* ap0 = nodes + (long)m0 * DIM + kk * 32 + (khalf << 3);
    const float* ap1 = nodes + (long)m1 * DIM + kk * 32 + (khalf << 3);
    v16bf a0 = frag_f32(ap0, ap0 + 16);
    v16bf a1 = frag_f32(ap1, ap1 + 16);
    v16bf b;
    int kb = kk * 32 + (khalf << 3);
#pragma unroll
    for (int e = 0; e < 16; ++e) b[e] = (__bf16)W[(kb + e + ((e >> 3) << 3)) * Nw + bc];
    acc0 = __builtin_amdgcn_wmma_f32_16x16x32_bf16(false, a0, false, b,
                                                   (short)0, acc0, false, false);
    acc1 = __builtin_amdgcn_wmma_f32_16x16x32_bf16(false, a1, false, b,
                                                   (short)0, acc1, false, false);
  }

  int col = nt * 16 + lrow;
  int cin = col & 511;
  int h   = cin >> 6;
  int d   = cin & 63;
  if (sec < 3) {                            // Q / K / V (wave-uniform branch)
    __bf16* dst = (sec == 0) ? Qb : (sec == 1) ? Kb : Vb;
#pragma unroll
    for (int half = 0; half < 2; ++half)
#pragma unroll
      for (int v = 0; v < 8; ++v) {
        int r   = mt2 * 32 + half * 16 + v + (khalf << 3);
        int bat = r >> 10;
        int i   = r & 1023;
        float val = half ? acc1[v] : acc0[v];
        dst[((long)(bat * HEADS + h) * NN + i) * DH + d] = (__bf16)val;
      }
  } else {                                  // gate = sigmoid
#pragma unroll
    for (int half = 0; half < 2; ++half)
#pragma unroll
      for (int v = 0; v < 8; ++v) {
        int r = mt2 * 32 + half * 16 + v + (khalf << 3);
        float val = half ? acc1[v] : acc0[v];
        gate[(long)r * INNER + cin] =
            __builtin_amdgcn_rcpf(1.0f + __expf(-val));
      }
  }
}

// ---------------------------------------------------------------------------
// Kernel 2: edge bias E[b,h,i,j] = (edges @ We), bf16 out, edges read once.
// Each wave: 8 consecutive 16-pair tiles, sharing hoisted We B-fragments.
// A loads are pure float4 streams; results packed into one b128 store/lane
// (8 consecutive j positions of the same (b,h,i) row, 16B aligned).
// ---------------------------------------------------------------------------
__global__ __launch_bounds__(256) void edge_bias(
    const float* __restrict__ edges, const float* __restrict__ We,
    __bf16* __restrict__ E)
{
  int  wave  = threadIdx.x >> 5;
  int  lane  = threadIdx.x & 31;
  long group = (long)blockIdx.x * 8 + wave;   // 16384 groups of 8 tiles
  int  khalf = lane >> 4;
  int  lrow  = lane & 15;

  v16bf bw0, bw1;                             // hoisted (We is 64x8)
#pragma unroll
  for (int e = 0; e < 16; ++e) {
    int k0 = (khalf << 3) + e + ((e >> 3) << 3);
    bw0[e] = (lrow < HEADS) ? (__bf16)We[k0 * HEADS + lrow]        : (__bf16)0.0f;
    bw1[e] = (lrow < HEADS) ? (__bf16)We[(32 + k0) * HEADS + lrow] : (__bf16)0.0f;
  }

  for (int t = 0; t < 8; ++t) {
    long p0 = (group * 8 + t) * 16;
    const float* ap = edges + (p0 + lrow) * 64 + (khalf << 3);
    v16bf a0 = frag_f32(ap,      ap + 16);
    v16bf a1 = frag_f32(ap + 32, ap + 48);
    v8f acc = {};
    acc = __builtin_amdgcn_wmma_f32_16x16x32_bf16(false, a0, false, bw0,
                                                  (short)0, acc, false, false);
    acc = __builtin_amdgcn_wmma_f32_16x16x32_bf16(false, a1, false, bw1,
                                                  (short)0, acc, false, false);
    if (lrow < HEADS) {                       // column = head
      union { uint4 u; __bf16 b[8]; } st;
#pragma unroll
      for (int v = 0; v < 8; ++v) st.b[v] = (__bf16)acc[v];
      long p   = p0 + (khalf << 3);           // first of 8 consecutive pairs
      long bat = p >> 20;
      int  i   = (int)((p >> 10) & 1023);
      int  j   = (int)(p & 1023);
      *(uint4*)(E + ((bat * HEADS + lrow) * (long)NN + i) * NN + j) = st.u;
    }
  }
}

// ---------------------------------------------------------------------------
// Kernel 3: flash attention. WG = (b, h, 64-row i-block); 4 waves x 16 rows.
// 256 workgroups. j-tiles of 64; K/V staged in LDS (V transposed);
// all LDS fragment reads are b128.
// ---------------------------------------------------------------------------
__global__ __launch_bounds__(128) void attn_kernel(
    const __bf16* __restrict__ Qb, const __bf16* __restrict__ Kb,
    const __bf16* __restrict__ Vb, const __bf16* __restrict__ E,
    const float* __restrict__ gate, __bf16* __restrict__ X)
{
  __shared__ __bf16 Ks[64][80];        // [j][d], row stride 160B (16B-aligned)
  __shared__ __bf16 VsT[64][80];       // [d][j] (transposed at fill time)
  __shared__ __bf16 Ps[4][16][80];     // per-wave P repack buffer

  int wave  = threadIdx.x >> 5;
  int lane  = threadIdx.x & 31;
  int blk   = blockIdx.x;              // 256 = 2b x 8h x 16 i-blocks
  int bat   = blk >> 7;
  int h     = (blk >> 4) & 7;
  int ib    = blk & 15;
  int i0    = ib * 64 + wave * 16;
  int khalf = lane >> 4;
  int lrow  = lane & 15;

  const __bf16* Qh = Qb + (long)(bat * HEADS + h) * NN * DH;
  const __bf16* Kh = Kb + (long)(bat * HEADS + h) * NN * DH;
  const __bf16* Vh = Vb + (long)(bat * HEADS + h) * NN * DH;
  const __bf16* Eh = E  + (long)(bat * HEADS + h) * NN * NN;

  // Q fragments (K = d = 64 -> two k-steps), b128 loads
  const __bf16* qp = Qh + (long)(i0 + lrow) * DH + (khalf << 3);
  v16bf aq0 = frag_bf16(qp,      qp + 16);
  v16bf aq1 = frag_bf16(qp + 32, qp + 48);

  float m[8], l[8];
  v8f   o[4];
#pragma unroll
  for (int v = 0; v < 8; ++v) { m[v] = -1e30f; l[v] = 0.0f; }
#pragma unroll
  for (int dt = 0; dt < 4; ++dt) o[dt] = (v8f){};

  for (int j0 = 0; j0 < NN; j0 += 64) {
    __syncthreads();
    {   // cooperative K/V tile load: 2 threads per j-row, 32 bf16 each
      int t  = threadIdx.x;
      int jr = t >> 1;
      int dc = (t & 1) * 32;
      const uint4* kg = (const uint4*)(Kh + (long)(j0 + jr) * DH + dc);
      uint4 k0 = kg[0], k1 = kg[1], k2 = kg[2], k3 = kg[3];
      *(uint4*)&Ks[jr][dc]      = k0;
      *(uint4*)&Ks[jr][dc + 8]  = k1;
      *(uint4*)&Ks[jr][dc + 16] = k2;
      *(uint4*)&Ks[jr][dc + 24] = k3;
      const __bf16* vg = Vh + (long)(j0 + jr) * DH + dc;
#pragma unroll
      for (int e = 0; e < 32; ++e) VsT[dc + e][jr] = vg[e];   // transpose
    }
    __syncthreads();
    if (j0 + 64 < NN) {
      __builtin_prefetch(Kh + (long)(j0 + 64) * DH, 0, 0);
      __builtin_prefetch(Vh + (long)(j0 + 64) * DH, 0, 0);
      __builtin_prefetch(Eh + (long)(i0 + (khalf << 3)) * NN + j0 + 64, 0, 0);
    }

    // ---- S = Q K^T * scale + bias -------------------------------------
    float s[4][8];
#pragma unroll
    for (int ct = 0; ct < 4; ++ct) {
      int jj = ct * 16 + lrow;                       // B column = key index
      const __bf16* kp = &Ks[jj][khalf << 3];
      v16bf bk0 = frag_bf16(kp,      kp + 16);
      v16bf bk1 = frag_bf16(kp + 32, kp + 48);
      v8f sa = {};
      sa = __builtin_amdgcn_wmma_f32_16x16x32_bf16(false, aq0, false, bk0,
                                                   (short)0, sa, false, false);
      sa = __builtin_amdgcn_wmma_f32_16x16x32_bf16(false, aq1, false, bk1,
                                                   (short)0, sa, false, false);
#pragma unroll
      for (int v = 0; v < 8; ++v) {
        int r = v + (khalf << 3);
        float bias = (float)Eh[(long)(i0 + r) * NN + j0 + ct * 16 + lrow];
        s[ct][v] = sa[v] * SCALE_F + bias;
      }
    }

    // ---- online softmax (row lives across 16 lanes @ same VGPR) -------
#pragma unroll
    for (int v = 0; v < 8; ++v) {
      float mx = s[0][v];
#pragma unroll
      for (int ct = 1; ct < 4; ++ct) mx = fmaxf(mx, s[ct][v]);
#pragma unroll
      for (int off = 1; off < 16; off <<= 1) mx = fmaxf(mx, __shfl_xor(mx, off, 32));
      float mn  = fmaxf(m[v], mx);
      float fac = __expf(m[v] - mn);
      m[v] = mn;
      l[v] *= fac;
#pragma unroll
      for (int dt = 0; dt < 4; ++dt) o[dt][v] *= fac;
      float rs = 0.0f;
#pragma unroll
      for (int ct = 0; ct < 4; ++ct) {
        float p = __expf(s[ct][v] - mn);
        s[ct][v] = p;
        rs += p;
      }
#pragma unroll
      for (int off = 1; off < 16; off <<= 1) rs += __shfl_xor(rs, off, 32);
      l[v] += rs;
    }

    // ---- repack P: C/D layout -> A layout via per-wave LDS ------------
#pragma unroll
    for (int ct = 0; ct < 4; ++ct)
#pragma unroll
      for (int v = 0; v < 8; ++v)
        Ps[wave][v + (khalf << 3)][ct * 16 + lrow] = (__bf16)s[ct][v];

    const __bf16* pp = &Ps[wave][lrow][khalf << 3];
    v16bf ap0 = frag_bf16(pp,      pp + 16);
    v16bf ap1 = frag_bf16(pp + 32, pp + 48);

    // ---- O += P V ------------------------------------------------------
#pragma unroll
    for (int dt = 0; dt < 4; ++dt) {
      int dd = dt * 16 + lrow;                      // B column = d index
      const __bf16* vp = &VsT[dd][khalf << 3];
      v16bf bv0 = frag_bf16(vp,      vp + 16);
      v16bf bv1 = frag_bf16(vp + 32, vp + 48);
      o[dt] = __builtin_amdgcn_wmma_f32_16x16x32_bf16(false, ap0, false, bv0,
                                                      (short)0, o[dt], false, false);
      o[dt] = __builtin_amdgcn_wmma_f32_16x16x32_bf16(false, ap1, false, bv1,
                                                      (short)0, o[dt], false, false);
    }
  }

  // epilogue: 1/l (single v_rcp), gate, store bf16 X (b*n, 512)
#pragma unroll
  for (int v = 0; v < 8; ++v) {
    float inv = __builtin_amdgcn_rcpf(l[v]);
    int   r   = i0 + v + (khalf << 3);
    long  base = ((long)bat * NN + r) * INNER + h * DH;
#pragma unroll
    for (int dt = 0; dt < 4; ++dt) {
      int d = dt * 16 + lrow;
      float val = o[dt][v] * inv * gate[base + d];
      X[base + d] = (__bf16)val;
    }
  }
}

// ---------------------------------------------------------------------------
// Kernel 4: out = X @ Wo  (M=2048, K=512, N=256), f32 output.
// Each wave: 32x16 output (2 M-tiles sharing B fragments).
// ---------------------------------------------------------------------------
__global__ __launch_bounds__(256) void out_proj(
    const __bf16* __restrict__ X, const float* __restrict__ Wo,
    float* __restrict__ out)
{
  int wave  = threadIdx.x >> 5;
  int lane  = threadIdx.x & 31;
  int tile  = blockIdx.x * 8 + wave;   // 1024 tiles = 64 mt2 x 16 nt
  int mt2   = tile >> 4;
  int nt    = tile & 15;
  int khalf = lane >> 4;
  int lrow  = lane & 15;
  int m0    = mt2 * 32 + lrow;
  int m1    = m0 + 16;
  int col   = nt * 16 + lrow;

  v8f acc0 = {}, acc1 = {};
  for (int kk = 0; kk < 16; ++kk) {
    const __bf16* ap0 = X + (long)m0 * INNER + kk * 32 + (khalf << 3);
    const __bf16* ap1 = X + (long)m1 * INNER + kk * 32 + (khalf << 3);
    v16bf a0 = frag_bf16(ap0, ap0 + 16);
    v16bf a1 = frag_bf16(ap1, ap1 + 16);
    v16bf b;
    int kb = kk * 32 + (khalf << 3);
#pragma unroll
    for (int e = 0; e < 16; ++e) b[e] = (__bf16)Wo[(kb + e + ((e >> 3) << 3)) * DIM + col];
    acc0 = __builtin_amdgcn_wmma_f32_16x16x32_bf16(false, a0, false, b,
                                                   (short)0, acc0, false, false);
    acc1 = __builtin_amdgcn_wmma_f32_16x16x32_bf16(false, a1, false, b,
                                                   (short)0, acc1, false, false);
  }
#pragma unroll
  for (int v = 0; v < 8; ++v) {
    int r0 = mt2 * 32 + v + (khalf << 3);
    out[(long)r0 * DIM + col]        = acc0[v];
    out[(long)(r0 + 16) * DIM + col] = acc1[v];
  }
}

// ---------------------------------------------------------------------------
extern "C" void kernel_launch(void* const* d_in, const int* in_sizes, int n_in,
                              void* d_out, int out_size, void* d_ws, size_t ws_size,
                              hipStream_t stream) {
  const float* nodes = (const float*)d_in[0];
  const float* edges = (const float*)d_in[1];
  // d_in[2] = mask (all-true in setup; no-op in reference) — ignored
  const float* Wq  = (const float*)d_in[3];
  const float* Wkv = (const float*)d_in[4];
  const float* We  = (const float*)d_in[5];
  const float* Wg  = (const float*)d_in[6];
  const float* Wo  = (const float*)d_in[7];

  char* ws = (char*)d_ws;                       // needs ~44 MB
  __bf16* Qb   = (__bf16*)(ws + (0ull  << 20));
  __bf16* Kb   = (__bf16*)(ws + (2ull  << 20));
  __bf16* Vb   = (__bf16*)(ws + (4ull  << 20));
  float*  gate = (float*) (ws + (6ull  << 20));
  __bf16* E    = (__bf16*)(ws + (10ull << 20));
  __bf16* X    = (__bf16*)(ws + (42ull << 20));

  qkvg_proj  <<<dim3(1024), dim3(256), 0, stream>>>(nodes, Wq, Wkv, Wg, Qb, Kb, Vb, gate);
  edge_bias  <<<dim3(2048), dim3(256), 0, stream>>>(edges, We, E);
  attn_kernel<<<dim3(256),  dim3(128), 0, stream>>>(Qb, Kb, Vb, E, gate, X);
  out_proj   <<<dim3(128),  dim3(256), 0, stream>>>(X, Wo, (float*)d_out);
}
